// DlrmSmall_7421703487501
// MI455X (gfx1250) — matrix operand order repaired and checked by reading, not compiled, not run on hardware
//
#include <hip/hip_runtime.h>
#include <hip/hip_bf16.h>

typedef __attribute__((ext_vector_type(16))) __bf16 bf16x16;
typedef __attribute__((ext_vector_type(8)))  __bf16 bf16x8;
typedef __attribute__((ext_vector_type(4)))  __bf16 bf16x4;
typedef __attribute__((ext_vector_type(8)))  float  f32x8;
typedef __attribute__((ext_vector_type(4)))  float  f32x4;

union BF16V { bf16x16 v; bf16x8 h[2]; };

#define BATCH      65536
#define NDENSE     13
#define NCOLS      39       // 13 dense + 26 sparse
#define VOCAB      1000000
#define EMBED      128
#define NFEAT      27
#define TOPIN      506
#define TROW       512      // padded top-MLP input row

// ---- fragment loaders -------------------------------------------------------
// A (16x32 bf16, MxK): lane holds row M=lane%16; K chunks {klo..klo+7, 16+klo..+7},
// klo = (lane<16 ? 0 : 8).  Two contiguous 16B loads.
// B (32x16 bf16, KxN): lane holds col N=lane%16; K chunks {klo..klo+7, 16+klo..+7},
// klo = (lane<16 ? 0 : 8).  Works on a column-major (NxKpad) weight copy.
static __device__ inline bf16x16 frag2(const __bf16* p0, const __bf16* p1) {
    BF16V f;
    f.h[0] = *(const bf16x8*)p0;
    f.h[1] = *(const bf16x8*)p1;
    return f.v;
}

// ---- weight convert + transpose: src (KxN f32) -> dst (N x Kpad bf16) -------
__global__ void prep_wt(const float* __restrict__ src, __bf16* __restrict__ dst,
                        int K, int N, int Kpad) {
    int i = blockIdx.x * 256 + threadIdx.x;
    if (i >= N * Kpad) return;
    int n = i / Kpad, k = i - n * Kpad;
    dst[i] = (k < K) ? (__bf16)src[(size_t)k * N + n] : (__bf16)0.f;
}

// ---- dense features -> bf16, padded 13 -> 32 --------------------------------
__global__ void prep_dense(const float* __restrict__ x, __bf16* __restrict__ A0) {
    int i = blockIdx.x * 256 + threadIdx.x;        // over BATCH*32
    int b = i >> 5, j = i & 31;
    A0[i] = (j < NDENSE) ? (__bf16)x[(size_t)b * NCOLS + j] : (__bf16)0.f;
}

// ---- WMMA GEMM: C[M x N] = relu(A[M x Kpad] * Wt^T + bias) ------------------
// A row-major bf16 (lda = Kpad), Wt is N x Kpad bf16 (column-major weights),
// bias f32[N], C row-major bf16 (ldc = N).
// Block = 8 waves (4 M x 2 N), block tile 128x128; wave tile 32x64 = 2x4 WMMA
// frags -> 8 WMMA per 12 b128 loads per K-step (vs 4:8 for a 32x32 tile).
__global__ __launch_bounds__(256)
void gemm_bias_relu(const __bf16* __restrict__ A,
                    const __bf16* __restrict__ Wt, int Kpad,
                    const float* __restrict__ bias,
                    __bf16* __restrict__ C, int N) {
    const int lane = threadIdx.x & 31;
    const int wave = threadIdx.x >> 5;
    const int wm = wave & 3;            // 4 waves along M
    const int wn = wave >> 2;           // 2 waves along N
    const int rowBase = blockIdx.y * 128 + wm * 32;
    const int colBase = blockIdx.x * 128 + wn * 64;
    const int ln16 = lane & 15;
    const int hi   = lane >> 4;
    const int klo  = hi * 8;
    const int kb   = hi * 16;

    const __bf16* aRow[2];
    aRow[0] = A + (size_t)(rowBase + ln16) * Kpad;
    aRow[1] = aRow[0] + (size_t)16 * Kpad;
    const __bf16* bCol[4];
    bCol[0] = Wt + (size_t)(colBase + ln16) * Kpad;
    bCol[1] = bCol[0] + (size_t)16 * Kpad;
    bCol[2] = bCol[0] + (size_t)32 * Kpad;
    bCol[3] = bCol[0] + (size_t)48 * Kpad;

    f32x8 acc[2][4] = {};
    for (int k0 = 0; k0 < Kpad; k0 += 32) {
        bf16x16 a[2], b[4];
        #pragma unroll
        for (int mi = 0; mi < 2; ++mi)
            a[mi] = frag2(aRow[mi] + k0 + klo, aRow[mi] + k0 + 16 + klo);
        #pragma unroll
        for (int ni = 0; ni < 4; ++ni)
            b[ni] = frag2(bCol[ni] + k0 + klo, bCol[ni] + k0 + 16 + klo);
        #pragma unroll
        for (int mi = 0; mi < 2; ++mi)
            #pragma unroll
            for (int ni = 0; ni < 4; ++ni)
                acc[mi][ni] = __builtin_amdgcn_wmma_f32_16x16x32_bf16(
                    false, a[mi], false, b[ni], (short)0, acc[mi][ni], false, false);
    }

    // Epilogue. C/D layout: VGPR r holds (M = tileBase + r + 8*hi, N = lane%16).
    #pragma unroll
    for (int ni = 0; ni < 4; ++ni) {
        int col = colBase + ni * 16 + ln16;
        float bj = bias[col];
        #pragma unroll
        for (int mi = 0; mi < 2; ++mi) {
            int row0 = rowBase + mi * 16 + hi * 8;
            #pragma unroll
            for (int r = 0; r < 8; ++r) {
                float v = acc[mi][ni][r] + bj;
                v = v > 0.f ? v : 0.f;
                C[(size_t)(row0 + r) * N + col] = (__bf16)v;
            }
        }
    }
}

// ---- interaction: gather embeddings, Gram 27x27 via WMMA, assemble t --------
// One wave per sample.  LDS stack: 32 rows x 128 bf16 per sample (rows 27..31 zero).
// Row 0 (h, already bf16) is DMA'd global->LDS with GLOBAL_LOAD_ASYNC_TO_LDS_B64
// (ASYNCcnt path, no VGPR data registers); embedding rows need f32->bf16 so they
// go through VGPR converts.
__global__ __launch_bounds__(256)
void interact(const float* __restrict__ x, const float* __restrict__ emb,
              const __bf16* __restrict__ h, __bf16* __restrict__ t) {
    __shared__ __bf16 smem[8][32][EMBED];     // 64 KB
    const int lane = threadIdx.x & 31;
    const int wave = threadIdx.x >> 5;
    const int s = blockIdx.x * 8 + wave;
    __bf16 (*st)[EMBED] = smem[wave];

    // row 0: async DMA of 8B per lane, global h row -> LDS (raw bf16 copy)
    {
        unsigned ldsa = (unsigned)(uintptr_t)(&st[0][lane * 4]);   // generic low 32b = LDS byte offset
        const __bf16* ga = h + (size_t)s * EMBED + lane * 4;
        asm volatile("global_load_async_to_lds_b64 %0, %1, off"
                     :: "v"(ldsa), "v"(ga) : "memory");
    }

    // rows 1..26: gathered embedding rows (f32 -> bf16)
    for (int r = 1; r <= 26; ++r) {
        int idx = (int)x[(size_t)s * NCOLS + NDENSE + (r - 1)];
        idx %= VOCAB;
        f32x4 f = *(const f32x4*)(emb + (size_t)idx * EMBED + lane * 4);
        bf16x4 o;
        o[0] = (__bf16)f[0]; o[1] = (__bf16)f[1];
        o[2] = (__bf16)f[2]; o[3] = (__bf16)f[3];
        *(bf16x4*)(&st[r][lane * 4]) = o;
    }
    // rows 27..31: zero pad
    {
        bf16x4 z = {};
        for (int r = 27; r < 32; ++r) *(bf16x4*)(&st[r][lane * 4]) = z;
    }

    // drain the async row-0 copy before reading the stack
    asm volatile("s_wait_asynccnt 0x0" ::: "memory");

    const int ln16 = lane & 15;
    const int hi   = lane >> 4;
    const int klo  = hi * 8;
    const int kb   = hi * 16;

    // Gram: G(32x32) = S(32x128) * S^T.  B = S^T means B-fragment cols are S rows,
    // so both A and B fragments are contiguous chunks of LDS stack rows.
    f32x8 c00 = {}, c01 = {}, c10 = {}, c11 = {};
    #pragma unroll
    for (int k0 = 0; k0 < EMBED; k0 += 32) {
        bf16x16 a0 = frag2(&st[ln16][k0 + klo],      &st[ln16][k0 + 16 + klo]);
        bf16x16 a1 = frag2(&st[16 + ln16][k0 + klo], &st[16 + ln16][k0 + 16 + klo]);
        bf16x16 b0 = frag2(&st[ln16][k0 + klo],      &st[ln16][k0 + 16 + klo]);
        bf16x16 b1 = frag2(&st[16 + ln16][k0 + klo], &st[16 + ln16][k0 + 16 + klo]);
        c00 = __builtin_amdgcn_wmma_f32_16x16x32_bf16(false, a0, false, b0, (short)0, c00, false, false);
        c01 = __builtin_amdgcn_wmma_f32_16x16x32_bf16(false, a0, false, b1, (short)0, c01, false, false);
        c10 = __builtin_amdgcn_wmma_f32_16x16x32_bf16(false, a1, false, b0, (short)0, c10, false, false);
        c11 = __builtin_amdgcn_wmma_f32_16x16x32_bf16(false, a1, false, b1, (short)0, c11, false, false);
    }

    // assemble t row: [h (128) | upper-tri Gram (378) | zero pad (6)]
    __bf16* tr = t + (size_t)s * TROW;
    *(bf16x4*)(tr + lane * 4) = *(const bf16x4*)(&st[0][lane * 4]);
    if (lane < 6) tr[TOPIN + lane] = (__bf16)0.f;

    f32x8 acc[2][2] = {c00, c01, c10, c11};
    #pragma unroll
    for (int mi = 0; mi < 2; ++mi) {
        #pragma unroll
        for (int ni = 0; ni < 2; ++ni) {
            int j = ni * 16 + ln16;
            #pragma unroll
            for (int r = 0; r < 8; ++r) {
                int i = mi * 16 + hi * 8 + r;
                if (i < NFEAT && j < NFEAT && j >= i) {
                    int p = i * NFEAT - (i * (i - 1)) / 2 + (j - i);
                    tr[EMBED + p] = (__bf16)acc[mi][ni][r];
                }
            }
        }
    }
}

// ---- final 256 -> 1 layer (no relu), fp32 weights, wave reduction -----------
__global__ __launch_bounds__(256)
void final_layer(const __bf16* __restrict__ u, const float* __restrict__ w,
                 const float* __restrict__ bias, float* __restrict__ out) {
    const int lane = threadIdx.x & 31;
    const int wave = threadIdx.x >> 5;
    const int s = blockIdx.x * 8 + wave;
    bf16x8 v = *(const bf16x8*)(u + (size_t)s * 256 + lane * 8);
    float sum = 0.f;
    #pragma unroll
    for (int i = 0; i < 8; ++i) sum += (float)v[i] * w[lane * 8 + i];
    #pragma unroll
    for (int off = 16; off; off >>= 1) sum += __shfl_xor(sum, off, 32);
    if (lane == 0) out[s] = sum + bias[0];
}

// ---- host orchestration -----------------------------------------------------
extern "C" void kernel_launch(void* const* d_in, const int* in_sizes, int n_in,
                              void* d_out, int out_size, void* d_ws, size_t ws_size,
                              hipStream_t stream) {
    const float* x   = (const float*)d_in[0];
    const float* emb = (const float*)d_in[1];
    const float* bw[3] = {(const float*)d_in[2], (const float*)d_in[3], (const float*)d_in[4]};
    const float* bb[3] = {(const float*)d_in[5], (const float*)d_in[6], (const float*)d_in[7]};
    const float* tw[5] = {(const float*)d_in[8], (const float*)d_in[9], (const float*)d_in[10],
                          (const float*)d_in[11], (const float*)d_in[12]};
    const float* tb[5] = {(const float*)d_in[13], (const float*)d_in[14], (const float*)d_in[15],
                          (const float*)d_in[16], (const float*)d_in[17]};
    float* out = (float*)d_out;
    char* ws = (char*)d_ws;
    const size_t MB = 1ull << 20;

    // lifetime-aliased workspace layout (~437 MB peak)
    __bf16* A0 = (__bf16*)(ws + 0);          // 4 MB   (dead after stage 1)
    __bf16* u2 = (__bf16*)(ws + 0);          // 128 MB (stage 6)
    __bf16* h1 = (__bf16*)(ws + 128 * MB);   // 64 MB  (dead after stage 2)
    __bf16* u3 = (__bf16*)(ws + 128 * MB);   // 64 MB  (stage 7)
    __bf16* h2 = (__bf16*)(ws + 192 * MB);   // 32 MB  (dead after stage 3)
    __bf16* u4 = (__bf16*)(ws + 192 * MB);   // 32 MB  (stage 8)
    __bf16* h3 = (__bf16*)(ws + 224 * MB);   // 16 MB
    __bf16* t  = (__bf16*)(ws + 240 * MB);   // 64 MB
    __bf16* u1 = (__bf16*)(ws + 304 * MB);   // 128 MB
    __bf16* wb = (__bf16*)(ws + 432 * MB);   // ~4.6 MB bf16 transposed weights

    // bf16 weight copies (N x Kpad)
    struct WDesc { const float* src; int K, N, Kpad; };
    WDesc wd[7] = {
        {bw[0],   13,  512,   32},
        {bw[1],  512,  256,  512},
        {bw[2],  256,  128,  256},
        {tw[0],  506, 1024,  512},
        {tw[1], 1024, 1024, 1024},
        {tw[2], 1024,  512, 1024},
        {tw[3],  512,  256,  512},
    };
    __bf16* wp[7];
    size_t off = 0;
    for (int i = 0; i < 7; ++i) {
        wp[i] = wb + off;
        off += (size_t)wd[i].N * wd[i].Kpad;
        int elems = wd[i].N * wd[i].Kpad;
        prep_wt<<<(elems + 255) / 256, 256, 0, stream>>>(wd[i].src, wp[i],
                                                         wd[i].K, wd[i].N, wd[i].Kpad);
    }

    // bottom MLP
    prep_dense<<<(BATCH * 32) / 256, 256, 0, stream>>>(x, A0);
    gemm_bias_relu<<<dim3( 512 / 128, BATCH / 128), 256, 0, stream>>>(A0, wp[0],   32, bb[0], h1,  512);
    gemm_bias_relu<<<dim3( 256 / 128, BATCH / 128), 256, 0, stream>>>(h1, wp[1],  512, bb[1], h2,  256);
    gemm_bias_relu<<<dim3( 128 / 128, BATCH / 128), 256, 0, stream>>>(h2, wp[2],  256, bb[2], h3,  128);

    // embedding gather + pairwise interaction + top-input assembly
    interact<<<BATCH / 8, 256, 0, stream>>>(x, emb, h3, t);

    // top MLP
    gemm_bias_relu<<<dim3(1024 / 128, BATCH / 128), 256, 0, stream>>>(t,  wp[3],  512, tb[0], u1, 1024);
    gemm_bias_relu<<<dim3(1024 / 128, BATCH / 128), 256, 0, stream>>>(u1, wp[4], 1024, tb[1], u2, 1024);
    gemm_bias_relu<<<dim3( 512 / 128, BATCH / 128), 256, 0, stream>>>(u2, wp[5], 1024, tb[2], u3,  512);
    gemm_bias_relu<<<dim3( 256 / 128, BATCH / 128), 256, 0, stream>>>(u3, wp[6],  512, tb[3], u4,  256);

    // final 256 -> 1
    final_layer<<<BATCH / 8, 256, 0, stream>>>(u4, tw[4], tb[4], out);
}
